// GNN_3659312136314
// MI455X (gfx1250) — matrix operand order repaired and checked by reading, not compiled, not run on hardware
//
#include <hip/hip_runtime.h>
#include <hip/hip_bf16.h>

typedef __attribute__((ext_vector_type(16))) _Float16 v16h;
typedef __attribute__((ext_vector_type(8)))  float    v8f;

#define HD   256
#define HD2  128
#define ACT_NONE 0
#define ACT_GELU 1

__device__ __forceinline__ float gelu_f(float x) {
  return 0.5f * x * (1.0f + erff(x * 0.70710678118654752440f));
}

// ---------------------------------------------------------------------------
// Weight repack: W[K x N] (f32, row-major) -> Wp in wave32 WMMA B-fragment
// order, f16. Fragment layout (contiguous k order used here, B 32x16 f16): lane holds one
// column (col = tn*16 + lane&15), halves j=0..15 -> K = kc + (lane>>4)*16 + j.
// Wp index: (((tn*KC + c)*32 + lane)*16 + j). One-time cost K*N elements;
// afterwards every lane loads its fragment as 2 contiguous b128s.
// ---------------------------------------------------------------------------
__global__ void k_packB(const float* __restrict__ W, _Float16* __restrict__ Wp,
                        int N, int KC) {
  const long long total = (long long)N * (KC * 32);
  long long t = (long long)blockIdx.x * blockDim.x + threadIdx.x;
  if (t >= total) return;
  const int j    = (int)(t & 15);
  long long r    = t >> 4;
  const int lane = (int)(r & 31);
  r >>= 5;
  const int c  = (int)(r % KC);
  const int tn = (int)(r / KC);
  const int krow = c * 32 + (lane >> 4) * 16 + j;
  const int col  = tn * 16 + (lane & 15);
  Wp[t] = (_Float16)W[(size_t)krow * N + col];
}

// ---------------------------------------------------------------------------
// WMMA GEMM: C[M x N] = act( [A0 | A1][M x K] @ W[K x N] + bias )
// A = per-row concat of A0 (k0 cols) and A1 (K-k0 cols).
// Each wave computes a 16(M) x 64(N) tile: one A fragment feeds 4 independent
// v_wmma_f32_16x16x32_f16 issues per k-chunk (A traffic /4, XDL pipelining).
// B fragments come pre-packed (f16) from k_packB. 8 waves/block.
// N % 64 == 0, K % 32 == 0, k0 % 32 == 0.
// ---------------------------------------------------------------------------
__global__ __launch_bounds__(256) void k_gemm(
    const float* __restrict__ A0, int lda0,
    const float* __restrict__ A1, int lda1, int k0,
    const _Float16* __restrict__ Wp, const float* __restrict__ bias,
    float* __restrict__ C, int ldc, int M, int N, int K, int act)
{
  const int KC     = K >> 5;
  const int tilesG = N >> 6;                      // 64-wide column groups
  const int tiles  = ((M + 15) >> 4) * tilesG;
  const int wave   = threadIdx.x >> 5;
  const int tile   = blockIdx.x * 8 + wave;
  if (tile >= tiles) return;                      // wave-uniform: EXEC stays full
  const int tm   = tile / tilesG;
  const int tg   = tile - tm * tilesG;
  const int lane = threadIdx.x & 31;
  const int half = lane >> 4;                     // 0: lanes 0-15, 1: lanes 16-31
  const int l15  = lane & 15;
  int arow = tm * 16 + l15; if (arow > M - 1) arow = M - 1;   // clamp for loads

  v8f acc0 = {}, acc1 = {}, acc2 = {}, acc3 = {};
  for (int c = 0; c < KC; ++c) {
    const int kc = c << 5;
    const float* Ar;
    if (kc < k0) Ar = A0 + (size_t)arow * lda0 + kc;
    else         Ar = A1 + (size_t)arow * lda1 + (kc - k0);
    // A 16x32 f16 fragment: lanes 0-15 rows, halves 0-15 -> K+0..15;
    // lanes 16-31 -> K+16..31 (matches packed B order).
    const int b0 = half << 4;
    v16h a;
    #pragma unroll
    for (int j = 0; j < 8; ++j) {
      a[j]     = (_Float16)Ar[b0 + j];
      a[8 + j] = (_Float16)Ar[b0 + 8 + j];
    }
    // 4 pre-packed B fragments: contiguous 32B per lane, coalesced.
    const _Float16* Wb = Wp + ((((size_t)(tg * 4) * KC + c) * 32 + lane) << 4);
    const size_t tstride = (size_t)KC * 32 * 16;  // halves per column tile
    const v16h bm0 = *(const v16h*)(Wb);
    const v16h bm1 = *(const v16h*)(Wb + tstride);
    const v16h bm2 = *(const v16h*)(Wb + 2 * tstride);
    const v16h bm3 = *(const v16h*)(Wb + 3 * tstride);
    if (c + 1 < KC) __builtin_prefetch(Wb + (32 * 16), 0, 1);
    acc0 = __builtin_amdgcn_wmma_f32_16x16x32_f16(false, a, false, bm0, (short)0, acc0, false, false);
    acc1 = __builtin_amdgcn_wmma_f32_16x16x32_f16(false, a, false, bm1, (short)0, acc1, false, false);
    acc2 = __builtin_amdgcn_wmma_f32_16x16x32_f16(false, a, false, bm2, (short)0, acc2, false, false);
    acc3 = __builtin_amdgcn_wmma_f32_16x16x32_f16(false, a, false, bm3, (short)0, acc3, false, false);
  }
  // C/D layout: VGPR r -> row r (lanes 0-15) / row 8+r (lanes 16-31), col = l15.
  v8f accs[4] = {acc0, acc1, acc2, acc3};
  #pragma unroll
  for (int u = 0; u < 4; ++u) {
    const int col = (tg * 4 + u) * 16 + l15;
    const float bv = bias ? bias[col] : 0.0f;
    #pragma unroll
    for (int r = 0; r < 8; ++r) {
      const int row = tm * 16 + (half << 3) + r;
      if (row < M) {
        float v = accs[u][r] + bv;
        if (act == ACT_GELU) v = gelu_f(v);
        C[(size_t)row * ldc + col] = v;
      }
    }
  }
}

// --------------------------- small / irregular kernels ---------------------
__global__ void k_zero(float* __restrict__ p, long long n) {
  long long t = (long long)blockIdx.x * blockDim.x + threadIdx.x;
  if (t < n) p[t] = 0.0f;
}

__global__ void k_embed(const int* __restrict__ t, const float* __restrict__ emb,
                        float* __restrict__ xw) {
  const int pos = blockIdx.x, f = threadIdx.x;
  xw[(size_t)pos * HD + f] = emb[(size_t)t[pos] * HD + f];
}

// 1D conv (kernel 2/3/4) + GELU + max-over-time. grid=(100,3), block=128.
__global__ void k_conv(const float* __restrict__ xw,
                       const float* __restrict__ w2, const float* __restrict__ b2,
                       const float* __restrict__ w3, const float* __restrict__ b3,
                       const float* __restrict__ w4, const float* __restrict__ b4,
                       float* __restrict__ feats) {
  const int o = blockIdx.x, ksi = blockIdx.y, ks = ksi + 2;
  const float* w  = (ksi == 0) ? w2 : (ksi == 1) ? w3 : w4;
  const float* bb = (ksi == 0) ? b2 : (ksi == 1) ? b3 : b4;
  const float* wo = w + (size_t)o * HD * ks;
  const int P = 512 - ks + 1;
  float mx = -1e30f;
  for (int pos = threadIdx.x; pos < P; pos += blockDim.x) {
    float acc = bb[o];
    for (int i = 0; i < HD; ++i) {
      const float* xp = xw + (size_t)pos * HD + i;
      const float* wp = wo + (size_t)i * ks;
      for (int k = 0; k < ks; ++k) acc += xp[(size_t)k * HD] * wp[k];
    }
    mx = fmaxf(mx, gelu_f(acc));
  }
  __shared__ float red[128];
  red[threadIdx.x] = mx; __syncthreads();
  for (int s = 64; s > 0; s >>= 1) {
    if (threadIdx.x < s) red[threadIdx.x] = fmaxf(red[threadIdx.x], red[threadIdx.x + s]);
    __syncthreads();
  }
  if (threadIdx.x == 0) feats[ksi * 100 + o] = red[0];
}

// p_text = gelu(ln(feats @ lw)). one block x 256.
__global__ void k_text_head(const float* __restrict__ feats, const float* __restrict__ lw,
                            const float* __restrict__ g, const float* __restrict__ b,
                            float* __restrict__ ptext) {
  const int j = threadIdx.x;
  float h = 0.0f;
  for (int f = 0; f < 300; ++f) h += feats[f] * lw[(size_t)f * HD + j];
  __shared__ float red[HD];
  red[j] = h; __syncthreads();
  for (int s = 128; s > 0; s >>= 1) { if (j < s) red[j] += red[j + s]; __syncthreads(); }
  const float mu = red[0] / 256.0f; __syncthreads();
  const float d = h - mu;
  red[j] = d * d; __syncthreads();
  for (int s = 128; s > 0; s >>= 1) { if (j < s) red[j] += red[j + s]; __syncthreads(); }
  const float var = red[0] / 256.0f;
  ptext[j] = gelu_f(d * rsqrtf(var + 1e-5f) * g[j] + b[j]);
}

// aq = (gelu(p_text@q1w+q1b)) @ q2w + q2b. one block x 128.
__global__ void k_aq(const float* __restrict__ ptext,
                     const float* __restrict__ q1w, const float* __restrict__ q1b,
                     const float* __restrict__ q2w, const float* __restrict__ q2b,
                     float* __restrict__ aq) {
  __shared__ float h1[HD2];
  const int j = threadIdx.x;
  float a = q1b[j];
  for (int i = 0; i < HD; ++i) a += ptext[i] * q1w[(size_t)i * HD2 + j];
  h1[j] = gelu_f(a); __syncthreads();
  float o = q2b[j];
  for (int i = 0; i < HD2; ++i) o += h1[i] * q2w[(size_t)i * HD2 + j];
  aq[j] = o;
}

__global__ void k_vec_lin_gelu(const float* __restrict__ in, const float* __restrict__ w,
                               const float* __restrict__ b, float* __restrict__ out,
                               int K, int N) {
  const int j = threadIdx.x;
  if (j >= N) return;
  float a = b[j];
  for (int i = 0; i < K; ++i) a += in[i] * w[(size_t)i * N + j];
  out[j] = gelu_f(a);
}

// cvec = c3b + qf @ c3w[0:256] + ef @ c3w[256:512]
__global__ void k_cvec(const float* __restrict__ qf, const float* __restrict__ ef,
                       const float* __restrict__ c3w, const float* __restrict__ c3b,
                       float* __restrict__ cvec) {
  const int j = threadIdx.x;
  float a = c3b[j];
  for (int i = 0; i < HD; ++i) a += qf[i] * c3w[(size_t)i * HD + j];
  for (int i = 0; i < HD; ++i) a += ef[i] * c3w[(size_t)(HD + i) * HD + j];
  cvec[j] = a;
}

__global__ void k_rowdot(const float* __restrict__ A, int lda, const float* __restrict__ v,
                         const float* __restrict__ bias, float* __restrict__ out,
                         int M, int K) {
  const int i = blockIdx.x * blockDim.x + threadIdx.x;
  if (i >= M) return;
  float a = bias ? bias[0] : 0.0f;
  const float* r = A + (size_t)i * lda;
  for (int k = 0; k < K; ++k) a += r[k] * v[k];
  out[i] = a;
}

// keyword scatter: one block per entry, 128 lanes = features.
__global__ void k_scatter_kw(const int* __restrict__ kidx, const int* __restrict__ m,
                             const float* __restrict__ kemb, const float* __restrict__ w,
                             float* __restrict__ xk) {
  const int i = blockIdx.x, f = threadIdx.x;
  const int ki = kidx[i];
  atomicAdd(&xk[(size_t)m[i] * HD2 + f], kemb[(size_t)ki * HD2 + f] * w[ki]);
}

__global__ void k_scatter_cnt(const int* __restrict__ seg, float* __restrict__ cnt, int n) {
  const int t = blockIdx.x * blockDim.x + threadIdx.x;
  if (t < n) atomicAdd(&cnt[seg[t]], 1.0f);
}

// x[i] = [author_emb[n[i]] (128) | xk_sum[i]/max(cnt,1) (128)]
__global__ void k_build_x(const int* __restrict__ n, const float* __restrict__ aemb,
                          const float* __restrict__ xk, const float* __restrict__ cnt,
                          float* __restrict__ x) {
  const int i = blockIdx.x, f = threadIdx.x;
  float v;
  if (f < HD2) v = aemb[(size_t)n[i] * HD2 + f];
  else         v = xk[(size_t)i * HD2 + (f - HD2)] / fmaxf(cnt[i], 1.0f);
  x[(size_t)i * HD + f] = v;
}

// edge scatter: one block per edge, 256 lanes = features (coalesced gather).
__global__ void k_scatter_edges(const int* __restrict__ src, const int* __restrict__ dst,
                                const float* __restrict__ h, float* __restrict__ agg) {
  const int j = blockIdx.x, f = threadIdx.x;
  atomicAdd(&agg[(size_t)dst[j] * HD + f], h[(size_t)src[j] * HD + f]);
}

__global__ void k_div_cnt(float* __restrict__ agg, const float* __restrict__ cnt) {
  const int i = blockIdx.x, f = threadIdx.x;
  agg[(size_t)i * HD + f] /= fmaxf(cnt[i], 1.0f);
}

// x[i] = ln(c[i] + x[i]) with per-row mean/var. one block per row.
__global__ void k_resid_ln(const float* __restrict__ c, float* __restrict__ x,
                           const float* __restrict__ g, const float* __restrict__ b) {
  const int i = blockIdx.x, f = threadIdx.x;
  const float v = c[(size_t)i * HD + f] + x[(size_t)i * HD + f];
  __shared__ float red[HD];
  red[f] = v; __syncthreads();
  for (int s = 128; s > 0; s >>= 1) { if (f < s) red[f] += red[f + s]; __syncthreads(); }
  const float mu = red[0] / 256.0f; __syncthreads();
  const float d = v - mu;
  red[f] = d * d; __syncthreads();
  for (int s = 128; s > 0; s >>= 1) { if (f < s) red[f] += red[f + s]; __syncthreads(); }
  const float var = red[0] / 256.0f;
  x[(size_t)i * HD + f] = d * rsqrtf(var + 1e-5f) * g[f] + b[f];
}

// ---------------------------------------------------------------------------
static inline void gemm(const float* A0, int lda0, const float* A1, int lda1, int k0,
                        const float* W, _Float16* Wp, const float* bias,
                        float* C, int ldc, int M, int N, int K, int act, hipStream_t s) {
  const int KC = K / 32;
  const long long pk = (long long)N * K;
  k_packB<<<(int)((pk + 255) / 256), 256, 0, s>>>(W, Wp, N, KC);
  const int tiles  = ((M + 15) / 16) * (N / 64);
  const int blocks = (tiles + 7) / 8;
  k_gemm<<<blocks, 256, 0, s>>>(A0, lda0, A1, lda1, k0, Wp, bias, C, ldc, M, N, K, act);
}

static inline void zero(float* p, long long n, hipStream_t s) {
  k_zero<<<(int)((n + 255) / 256), 256, 0, s>>>(p, n);
}

extern "C" void kernel_launch(void* const* d_in, const int* in_sizes, int n_in,
                              void* d_out, int out_size, void* d_ws, size_t ws_size,
                              hipStream_t stream) {
  // --- inputs (setup_inputs dict order, params flattened in insertion order) ---
  const int* t_tok = (const int*)d_in[0];
  const int* nidx  = (const int*)d_in[1];
  const int* k_idx = (const int*)d_in[2];
  const int* mseg  = (const int*)d_in[3];
  const int* e0    = (const int*)d_in[4];
  const int* e1    = (const int*)d_in[5];
  const float* author_emb  = (const float*)d_in[6];
  const float* keyword_emb = (const float*)d_in[7];
  const float* cemb = (const float*)d_in[8];
  const float* c2w = (const float*)d_in[9],  *c2b = (const float*)d_in[10];
  const float* c3w = (const float*)d_in[11], *c3b = (const float*)d_in[12];
  const float* c4w = (const float*)d_in[13], *c4b = (const float*)d_in[14];
  const float* lw  = (const float*)d_in[15];
  const float* cg  = (const float*)d_in[16], *cbv = (const float*)d_in[17];
  const float* q1w = (const float*)d_in[18], *q1b = (const float*)d_in[19];
  const float* k1w = (const float*)d_in[20], *k1b = (const float*)d_in[21];
  const float* q2w = (const float*)d_in[22], *q2b = (const float*)d_in[23];
  const float* k2w = (const float*)d_in[24], *k2b = (const float*)d_in[25];
  const float* cp[2][10];
  for (int l = 0; l < 2; ++l)
    for (int i = 0; i < 10; ++i) cp[l][i] = (const float*)d_in[26 + l * 10 + i];
  const float* cls_c0w = (const float*)d_in[46], *cls_c0b = (const float*)d_in[47];
  const float* cls_c1w = (const float*)d_in[48], *cls_c1b = (const float*)d_in[49];
  const float* cls_c2w = (const float*)d_in[50], *cls_c2b = (const float*)d_in[51];
  const float* cls_c3w = (const float*)d_in[52], *cls_c3b = (const float*)d_in[53];
  const float* cls_c4w = (const float*)d_in[54], *cls_c4b = (const float*)d_in[55];
  const float* cls_c5w = (const float*)d_in[56], *cls_c5b = (const float*)d_in[57];

  const int N0 = 50000, N1 = 20000;
  const int Kkw = in_sizes[2];              // 250000
  const int E0  = in_sizes[4] / 2;          // 600000
  const int E1  = in_sizes[5] / 2;          // 240000
  const int NKW = in_sizes[7] / HD2;        // 20000

  // --- workspace arena (floats; base and all offsets 128B-aligned) ---
  float* W = (float*)d_ws;
  const size_t ROW = (size_t)N0 * HD;       // 12.8M floats
  float* x    = W;
  float* h1   = W + ROW;
  float* h2   = W + 2 * ROW;
  float* agg  = W + 3 * ROW;                // also reused as xk_sum (N0 x 128)
  float* cnt  = W + 4 * ROW;                // N0
  float* xw   = cnt + 50048;                // 512 x 256
  float* feats = xw + 512 * HD;             // 300 (+pad)
  float* ptext = feats + 512;
  float* aqv   = ptext + HD;
  float* qf    = aqv + HD2;
  float* efv   = qf + HD;
  float* cvec  = efv + HD;
  _Float16* wp = (_Float16*)(cvec + HD);    // packed weights, up to 512x256 f16

  float* pred = (float*)d_out;              // 19999
  float* wout = pred + (N1 - 1);            // 20000

  // ---------------- Stage 1: text CNN head ----------------
  k_embed<<<512, 256, 0, stream>>>(t_tok, cemb, xw);
  k_conv<<<dim3(100, 3), 128, 0, stream>>>(xw, c2w, c2b, c3w, c3b, c4w, c4b, feats);
  k_text_head<<<1, 256, 0, stream>>>(feats, lw, cg, cbv, ptext);
  k_aq<<<1, 128, 0, stream>>>(ptext, q1w, q1b, q2w, q2b, aqv);
  k_vec_lin_gelu<<<1, 256, 0, stream>>>(ptext, cls_c0w, cls_c0b, qf, HD, HD);

  // ---------------- Stage 2: keyword attention (WMMA) ----------------
  gemm(keyword_emb, HD2, nullptr, 0, HD2, k1w, wp, k1b, h1, HD2, NKW, HD2, HD2, ACT_GELU, stream);
  gemm(h1, HD2, nullptr, 0, HD2, k2w, wp, k2b, h2, HD2, NKW, HD2, HD2, ACT_NONE, stream);
  k_rowdot<<<(NKW + 255) / 256, 256, 0, stream>>>(h2, HD2, aqv, nullptr, wout, NKW, HD2);

  // ---------------- Stage 3: keyword segment-mean -> node features ----------
  zero(agg, (long long)N0 * HD2, stream);
  zero(cnt, N0, stream);
  k_scatter_kw<<<Kkw, 128, 0, stream>>>(k_idx, mseg, keyword_emb, wout, agg);
  k_scatter_cnt<<<(Kkw + 255) / 256, 256, 0, stream>>>(mseg, cnt, Kkw);
  k_build_x<<<N0, 256, 0, stream>>>(nidx, author_emb, agg, cnt, x);

  // ---------------- Stage 4: two graph convolutions ----------------
  for (int layer = 0; layer < 2; ++layer) {
    const int  sz   = (layer == 0) ? N0 : N1;
    const int  E    = (layer == 0) ? E0 : E1;
    const int* eptr = (layer == 0) ? e0 : e1;
    const float *l0w = cp[layer][0], *l0b = cp[layer][1];
    const float *l1w = cp[layer][2], *l1b = cp[layer][3];
    const float *l2w = cp[layer][4], *l2b = cp[layer][5];
    const float *l3w = cp[layer][6], *l3b = cp[layer][7];
    const float *gg  = cp[layer][8], *gb  = cp[layer][9];

    gemm(x,  HD, nullptr, 0, HD, l0w, wp, l0b, h1, HD, N0, HD, HD, ACT_GELU, stream);
    gemm(h1, HD, nullptr, 0, HD, l1w, wp, l1b, h2, HD, N0, HD, HD, ACT_GELU, stream);
    zero(agg, (long long)sz * HD, stream);
    zero(cnt, sz, stream);
    k_scatter_edges<<<E, 256, 0, stream>>>(eptr, eptr + E, h2, agg);
    k_scatter_cnt<<<(E + 255) / 256, 256, 0, stream>>>(eptr + E, cnt, E);
    k_div_cnt<<<sz, 256, 0, stream>>>(agg, cnt);
    // c = gelu([tgt|agg] @ l2w + b) — concat GEMM, K = 512
    gemm(x, HD, agg, HD, HD, l2w, wp, l2b, h1, HD, sz, HD, 2 * HD, ACT_GELU, stream);
    gemm(h1, HD, nullptr, 0, HD, l3w, wp, l3b, h2, HD, sz, HD, HD, ACT_GELU, stream);
    k_resid_ln<<<sz, 256, 0, stream>>>(h2, x, gg, gb);
  }

  // ---------------- Stage 5: classifier ----------------
  const int M = N1;                         // x now has N1 rows
  k_vec_lin_gelu<<<1, 256, 0, stream>>>(x, cls_c1w, cls_c1b, efv, HD, HD);  // ef from x[0]
  k_cvec<<<1, 256, 0, stream>>>(qf, efv, cls_c3w, cls_c3b, cvec);
  // et = gelu(x[1:] @ c2w + b)
  gemm(x + HD, HD, nullptr, 0, HD, cls_c2w, wp, cls_c2b, h1, HD, M - 1, HD, HD, ACT_GELU, stream);
  // h = gelu(et @ c3w[512:768] + cvec)   (qf/ef contributions folded into cvec)
  gemm(h1, HD, nullptr, 0, HD, cls_c3w + (size_t)512 * HD, wp, cvec, h2, HD, M - 1, HD, HD, ACT_GELU, stream);
  // h = gelu(h @ c4w + b)
  gemm(h2, HD, nullptr, 0, HD, cls_c4w, wp, cls_c4b, h1, HD2, M - 1, HD2, HD, ACT_GELU, stream);
  // pred = h @ c5w + c5b
  k_rowdot<<<(M - 1 + 255) / 256, 256, 0, stream>>>(h1, HD2, cls_c5w, cls_c5b, pred, M - 1, HD2);

  (void)n_in; (void)out_size; (void)ws_size;
}
